// BidirectionalCrossAttention_35931696398392
// MI455X (gfx1250) — matrix operand order repaired and checked by reading, not compiled, not run on hardware
//
#include <hip/hip_runtime.h>

// ---------------------------------------------------------------------------
// BidirectionalCrossAttention for MI455X (gfx1250, wave32, WMMA + async LDS)
//
//   out         = softmax_rows(S)   @ CV ; S = (x@W_qk)(ctx@W_cqk)^T * d^-1/2
//   context_out = softmax_rows(S^T) @ V
// 4x proj GEMM (f32->f16), 2x flash attention, 2x out GEMM (f16xW + bias).
// Operand tiles are double-buffered in LDS via GLOBAL_LOAD_ASYNC_TO_LDS_*
// (issue-early / wait-late, ASYNCcnt), feeding v_wmma_f32_16x16x32_f16.
// Softmax row reductions use DPP row_ror butterflies (pure VALU, no LDS).
// ---------------------------------------------------------------------------

typedef __attribute__((ext_vector_type(16))) _Float16 v16h;
typedef __attribute__((ext_vector_type(8)))  _Float16 v8h;
typedef __attribute__((ext_vector_type(8)))  float    v8f;

#define HEADS  16
#define DHEAD  64
#define DIM    1024
#define INNER  1024   // HEADS * DHEAD
#define SEQ    2048

__device__ __forceinline__ v8f wmma_f32_f16(v16h a, v16h b, v8f c) {
  // (neg_a, A, neg_b, B, c_mod, C, reuse_a, reuse_b)
  return __builtin_amdgcn_wmma_f32_16x16x32_f16(false, a, false, b,
                                                (short)0, c, false, false);
}

// ---- LDS helpers -----------------------------------------------------------
typedef __attribute__((address_space(3))) char lds_char;

__device__ __forceinline__ unsigned lds_addr_of(const void* p) {
  // generic -> local addrspace cast; local pointers are 32-bit LDS offsets
  return (unsigned)(unsigned long long)(lds_char*)(char*)p;
}

// async copies global -> LDS (per-lane addresses), tracked by ASYNCcnt
__device__ __forceinline__ void async_ld_b128(void* lds_dst, const void* gsrc) {
  unsigned           loff = lds_addr_of(lds_dst);
  unsigned long long ga   = (unsigned long long)gsrc;
  asm volatile("global_load_async_to_lds_b128 %0, %1, off"
               :: "v"(loff), "v"(ga) : "memory");
}
__device__ __forceinline__ void async_ld_b32(void* lds_dst, const void* gsrc) {
  unsigned           loff = lds_addr_of(lds_dst);
  unsigned long long ga   = (unsigned long long)gsrc;
  asm volatile("global_load_async_to_lds_b32 %0, %1, off"
               :: "v"(loff), "v"(ga) : "memory");
}
__device__ __forceinline__ void wait_async0() {
  asm volatile("s_wait_asynccnt 0" ::: "memory");
}

// ---- 16-lane row reductions via DPP row rotates (VALU only) ----------------
#define DPP_ROR(v, n)                                                          \
  __builtin_bit_cast(float, __builtin_amdgcn_update_dpp(                       \
      0, __builtin_bit_cast(int, (v)), 0x120 | (n), 0xf, 0xf, true))

__device__ __forceinline__ float row16_max(float v) {
  v = fmaxf(v, DPP_ROR(v, 1));
  v = fmaxf(v, DPP_ROR(v, 2));
  v = fmaxf(v, DPP_ROR(v, 4));
  v = fmaxf(v, DPP_ROR(v, 8));
  return v;
}
__device__ __forceinline__ float row16_sum(float v) {
  v += DPP_ROR(v, 1);
  v += DPP_ROR(v, 2);
  v += DPP_ROR(v, 4);
  v += DPP_ROR(v, 8);
  return v;
}

// ---- fragment loaders (contiguous-K order used by this file, A/B alike, wave32) ---------------
// A fragment (16x32 f16): lane holds row M = lane&15.
//   e in [0,16): K = 2*loff8 + e      (loff8 = (lane>=16) ? 8 : 0)
__device__ __forceinline__ v16h load_a_f32(const float* base, int loff8) {
  const float* p0 = base + 2 * loff8;
  const float* p1 = base + 2 * loff8 + 8;
  v16h a;
#pragma unroll
  for (int e = 0; e < 8; ++e) {
    a[e]     = (_Float16)p0[e];
    a[8 + e] = (_Float16)p1[e];
  }
  return a;
}

__device__ __forceinline__ v16h load_a_f16(const _Float16* base, int loff8) {
  v8h lo = *(const v8h*)(base + 2 * loff8);
  v8h hi = *(const v8h*)(base + 2 * loff8 + 8);
  v16h a;
#pragma unroll
  for (int e = 0; e < 8; ++e) { a[e] = lo[e]; a[8 + e] = hi[e]; }
  return a;
}

// B fragment (32x16 f16): lane holds col N = lane&15; element e -> K = koff16+e.
// base points at 16 contiguous f16 along K (already offset by koff16).
__device__ __forceinline__ v16h load_b_f16(const _Float16* base) {
  v8h lo = *(const v8h*)(base);
  v8h hi = *(const v8h*)(base + 8);
  v16h b;
#pragma unroll
  for (int e = 0; e < 8; ++e) { b[e] = lo[e]; b[8 + e] = hi[e]; }
  return b;
}

// ---------------------------------------------------------------------------
// GEMM: Out[M x N] = A[M x K] @ W_f32[K x N] (+ bias)
// Block: 8 waves -> 128x64 output tile. W chunk tile staged to LDS transposed
// (Wt[n][k], f32) via async b32 scatter; double-buffered so staging overlaps
// compute. B fragment = 16 contiguous f32 per lane (4x ds_load_b128).
// ---------------------------------------------------------------------------
#define WPAD 36   // row stride in dwords: 16B-aligned, odd*4 -> conflict-free

template<bool A_HALF, bool TRANS_OUT, bool F32_OUT>
__global__ __launch_bounds__(256)
void bca_gemm(const void* __restrict__ Av, const float* __restrict__ W,
              const float* __restrict__ bias,
              _Float16* __restrict__ OutH, float* __restrict__ OutF,
              int M, int N, int K)
{
  __shared__ float Wt[2][64][WPAD];   // [buf][n][k] transposed, 2x9KB

  const int tid  = threadIdx.x;
  const int wave = tid >> 5;
  const int lane = tid & 31;
  const int half = lane >> 4;
  const int lcol = lane & 15;
  const int loff8  = half * 8;
  const int koff16 = half * 16;

  const int row0 = blockIdx.x * 128 + wave * 16;  // output rows for this wave
  const int n0   = blockIdx.y * 64;               // output cols for this block
  const int arow = row0 + lcol;

  const float*    __restrict__ Af = (const float*)Av;
  const _Float16* __restrict__ Ah = (const _Float16*)Av;

  // staging indices: 2048 elements / 256 threads = 8 each; coalesced along n
  const int sn = tid & 63;          // column within [n0, n0+64)
  const int sr = tid >> 6;          // base k row (step 4)

  v8f acc[4] = {};
  const int NC = K / 32;

  // prologue: stage chunk 0 into buf 0
#pragma unroll
  for (int it = 0; it < 8; ++it) {
    const int r = sr + it * 4;
    async_ld_b32(&Wt[0][sn][r], W + (size_t)r * N + n0 + sn);
  }

  for (int c = 0; c < NC; ++c) {
    const int buf = c & 1;
    wait_async0();
    __syncthreads();                 // staged buf visible; other buf free

    if (c + 1 < NC) {                // issue next chunk early (overlaps WMMA)
      const int k1 = (c + 1) * 32;
#pragma unroll
      for (int it = 0; it < 8; ++it) {
        const int r = sr + it * 4;
        async_ld_b32(&Wt[buf ^ 1][sn][r], W + (size_t)(k1 + r) * N + n0 + sn);
      }
    }

    // ---- A fragment (one per wave per chunk, reused by 4 WMMAs) ----
    const int k0 = c * 32;
    v16h a;
    if constexpr (A_HALF) {
      a = load_a_f16(Ah + (size_t)arow * K + k0, loff8);
    } else {
      a = load_a_f32(Af + (size_t)arow * K + k0, loff8);
    }

    // ---- 4 N-tiles: B frags = contiguous f32 runs from transposed LDS ----
#pragma unroll
    for (int nt = 0; nt < 4; ++nt) {
      const float* wp = &Wt[buf][nt * 16 + lcol][koff16];
      v16h b;
#pragma unroll
      for (int e = 0; e < 16; ++e) b[e] = (_Float16)wp[e];
      acc[nt] = wmma_f32_f16(a, b, acc[nt]);
    }
  }

  // ---- store (C-layout: lane holds col, vgpr r holds row r + 8*half) ----
#pragma unroll
  for (int nt = 0; nt < 4; ++nt) {
    const int ocol = n0 + nt * 16 + lcol;
#pragma unroll
    for (int r = 0; r < 8; ++r) {
      const int orow = row0 + r + 8 * half;
      if constexpr (F32_OUT) {
        OutF[(size_t)orow * N + ocol] = acc[nt][r] + bias[ocol];
      } else if constexpr (TRANS_OUT) {
        OutH[(size_t)ocol * M + orow] = (_Float16)acc[nt][r];
      } else {
        OutH[(size_t)orow * N + ocol] = (_Float16)acc[nt][r];
      }
    }
  }
}

// ---------------------------------------------------------------------------
// flash_attn: Out_f16[SEQ x INNER] = softmax_rows(Q K^T * scale) @ V  per head
//   Q, Km : f16 [SEQ][INNER] row-major ; Vt : f16 [INNER][SEQ] (transposed)
// 8 waves / block share head h and march j in lockstep; K/V chunk tiles are
// double-buffered in LDS via async-to-LDS b128 (8x reuse, overlapped).
// ---------------------------------------------------------------------------
__global__ __launch_bounds__(256)
void bca_flash_attn(const _Float16* __restrict__ Q,
                    const _Float16* __restrict__ Km,
                    const _Float16* __restrict__ Vt,
                    _Float16* __restrict__ Out,
                    float scale)
{
  __shared__ _Float16 Klds[2][32][72];    // 32 j-rows x 64 d (+8 pad)
  __shared__ _Float16 Vlds[2][64][40];    // 64 d-rows x 32 j (+8 pad)
  __shared__ _Float16 Plds[8][16][32];    // per-wave probability tile

  const int tid  = threadIdx.x;
  const int wave = tid >> 5;
  const int lane = tid & 31;
  const int half = lane >> 4;
  const int lcol = lane & 15;
  const int loff8  = half * 8;
  const int koff16 = half * 16;

  const int h  = blockIdx.y;
  const int i0 = (blockIdx.x * 8 + wave) * 16;

  // Q fragments (A-layout, loaded once per wave); fold softmax scale into Q
  const _Float16* __restrict__ qrow = Q + (size_t)(i0 + lcol) * INNER + h * DHEAD;
  v16h q0 = load_a_f16(qrow,      loff8);
  v16h q1 = load_a_f16(qrow + 32, loff8);
  const _Float16 hs = (_Float16)scale;    // 0.125: exact in f16
#pragma unroll
  for (int e = 0; e < 16; ++e) { q0[e] *= hs; q1[e] *= hs; }

  float m[8], l[8];
  v8f o0 = {}, o1 = {}, o2 = {}, o3 = {};
#pragma unroll
  for (int r = 0; r < 8; ++r) { m[r] = -__builtin_inff(); l[r] = 0.0f; }

  // per-thread staging indices
  const int kr = tid >> 3, ks = tid & 7;   // K: 32 rows x 8 (16B) segs
  const int vr = tid >> 2, vs = tid & 3;   // V: 64 rows x 4 (16B) segs
  const _Float16* __restrict__ ksrc = Km + (size_t)kr * INNER + h * DHEAD + ks * 8;
  const _Float16* __restrict__ vsrc = Vt + (size_t)(h * DHEAD + vr) * SEQ + vs * 8;

  // prologue: stage chunk 0 into buf 0
  async_ld_b128(&Klds[0][kr][ks * 8], ksrc);
  async_ld_b128(&Vlds[0][vr][vs * 8], vsrc);

  const int NC = SEQ / 32;
  for (int c = 0; c < NC; ++c) {
    const int buf = c & 1;
    wait_async0();
    __syncthreads();                 // staged buf visible; other buf free

    if (c + 1 < NC) {                // issue next chunk early (overlaps WMMA)
      const size_t j1 = (size_t)(c + 1) * 32;
      async_ld_b128(&Klds[buf ^ 1][kr][ks * 8], ksrc + j1 * INNER);
      async_ld_b128(&Vlds[buf ^ 1][vr][vs * 8], vsrc + j1);
    }

    // ---- S = Q @ K^T for this 16x32 chunk (two 16x16 tiles) ----
    const _Float16* kp0 = &Klds[buf][lcol][koff16];
    const _Float16* kp1 = &Klds[buf][16 + lcol][koff16];
    v8f s0 = {}, s1 = {};
    s0 = wmma_f32_f16(q0, load_b_f16(kp0),      s0);
    s0 = wmma_f32_f16(q1, load_b_f16(kp0 + 32), s0);
    s1 = wmma_f32_f16(q0, load_b_f16(kp1),      s1);
    s1 = wmma_f32_f16(q1, load_b_f16(kp1 + 32), s1);

    // ---- online softmax: DPP row_ror butterflies across the 16-lane half --
    float alpha[8];
#pragma unroll
    for (int r = 0; r < 8; ++r) {
      const float rmax = row16_max(fmaxf(s0[r], s1[r]));
      const float mn   = fmaxf(m[r], rmax);
      alpha[r] = __expf(m[r] - mn);
      m[r] = mn;
      s0[r] = __expf(s0[r] - mn);
      s1[r] = __expf(s1[r] - mn);
      const float rsum = row16_sum(s0[r] + s1[r]);
      l[r] = l[r] * alpha[r] + rsum;
    }
#pragma unroll
    for (int r = 0; r < 8; ++r) {
      o0[r] *= alpha[r]; o1[r] *= alpha[r];
      o2[r] *= alpha[r]; o3[r] *= alpha[r];
    }

    // ---- transpose P (C-layout) -> A-layout via per-wave LDS tile ----
#pragma unroll
    for (int r = 0; r < 8; ++r) {
      Plds[wave][r + 8 * half][lcol]      = (_Float16)s0[r];
      Plds[wave][r + 8 * half][16 + lcol] = (_Float16)s1[r];
    }
    v16h p;
    {
      const _Float16* prow = &Plds[wave][lcol][0];
#pragma unroll
      for (int t = 0; t < 8; ++t) {
        const int kk = 2 * t + 2 * loff8;  // even
        const unsigned pk = *(const unsigned*)(prow + kk);
        p[2 * t]     = __builtin_bit_cast(_Float16, (unsigned short)(pk & 0xFFFFu));
        p[2 * t + 1] = __builtin_bit_cast(_Float16, (unsigned short)(pk >> 16));
      }
    }

    // ---- O += P @ V  (4 N-tiles of 16 over head dim, B frags from LDS) ----
    o0 = wmma_f32_f16(p, load_b_f16(&Vlds[buf][0 * 16 + lcol][koff16]), o0);
    o1 = wmma_f32_f16(p, load_b_f16(&Vlds[buf][1 * 16 + lcol][koff16]), o1);
    o2 = wmma_f32_f16(p, load_b_f16(&Vlds[buf][2 * 16 + lcol][koff16]), o2);
    o3 = wmma_f32_f16(p, load_b_f16(&Vlds[buf][3 * 16 + lcol][koff16]), o3);
  }

  // ---- normalize and store (C-layout scatter) ----
#pragma unroll
  for (int r = 0; r < 8; ++r) {
    const float inv = 1.0f / l[r];
    _Float16* op = Out + (size_t)(i0 + r + 8 * half) * INNER + h * DHEAD + lcol;
    op[0]  = (_Float16)(o0[r] * inv);
    op[16] = (_Float16)(o1[r] * inv);
    op[32] = (_Float16)(o2[r] * inv);
    op[48] = (_Float16)(o3[r] * inv);
  }
}

// ---------------------------------------------------------------------------
// launch
// ---------------------------------------------------------------------------
extern "C" void kernel_launch(void* const* d_in, const int* in_sizes, int n_in,
                              void* d_out, int out_size, void* d_ws, size_t ws_size,
                              hipStream_t stream) {
  (void)in_sizes; (void)n_in; (void)out_size; (void)ws_size;

  const float* x       = (const float*)d_in[0];
  const float* context = (const float*)d_in[1];
  const float* W_qk    = (const float*)d_in[2];
  const float* W_cqk   = (const float*)d_in[3];
  const float* W_v     = (const float*)d_in[4];
  const float* W_cv    = (const float*)d_in[5];
  const float* W_out   = (const float*)d_in[6];
  const float* b_out   = (const float*)d_in[7];
  const float* W_cout  = (const float*)d_in[8];
  const float* b_cout  = (const float*)d_in[9];

  _Float16* ws = (_Float16*)d_ws;
  const size_t NP = (size_t)SEQ * INNER;   // 2M halves per tensor
  _Float16* QK  = ws + 0 * NP;             // [SEQ][INNER]
  _Float16* CQK = ws + 1 * NP;             // [SEQ][INNER]
  _Float16* Vt  = ws + 2 * NP;             // [INNER][SEQ] (transposed)
  _Float16* CVt = ws + 3 * NP;             // [INNER][SEQ] (transposed)
  _Float16* Oa  = ws + 4 * NP;             // attn output (x side), f16
  _Float16* Ob  = ws + 5 * NP;             // attn output (context side), f16

  float* out0 = (float*)d_out;             // out          [SEQ][DIM]
  float* out1 = out0 + (size_t)SEQ * DIM;  // context_out  [SEQ][DIM]

  const dim3 blk(256);
  const dim3 ggrid(SEQ / 128, INNER / 64); // (16,16)
  const float scale = 0.125f;              // DIM_HEAD^-0.5

  // input projections (f32 A, f16 out; V/CV stored transposed)
  bca_gemm<false, false, false><<<ggrid, blk, 0, stream>>>(x,       W_qk,  nullptr, QK,  nullptr, SEQ, INNER, DIM);
  bca_gemm<false, false, false><<<ggrid, blk, 0, stream>>>(context, W_cqk, nullptr, CQK, nullptr, SEQ, INNER, DIM);
  bca_gemm<false, true,  false><<<ggrid, blk, 0, stream>>>(x,       W_v,   nullptr, Vt,  nullptr, SEQ, INNER, DIM);
  bca_gemm<false, true,  false><<<ggrid, blk, 0, stream>>>(context, W_cv,  nullptr, CVt, nullptr, SEQ, INNER, DIM);

  // bidirectional attention: same kernel, swapped operands
  const dim3 fgrid(SEQ / 16 / 8, HEADS);   // (16,16)
  bca_flash_attn<<<fgrid, blk, 0, stream>>>(QK,  CQK, CVt, Oa, scale);
  bca_flash_attn<<<fgrid, blk, 0, stream>>>(CQK, QK,  Vt,  Ob, scale);

  // output projections (f16 A, f32 W + bias, f32 out)
  bca_gemm<true, false, true><<<ggrid, blk, 0, stream>>>(Oa, W_out,  b_out,  nullptr, out0, SEQ, DIM, INNER);
  bca_gemm<true, false, true><<<ggrid, blk, 0, stream>>>(Ob, W_cout, b_cout, nullptr, out1, SEQ, DIM, INNER);
}
